// FeatureSpaceRegionScore_4166118277842
// MI455X (gfx1250) — matrix operand order repaired and checked by reading, not compiled, not run on hardware
//
#include <hip/hip_runtime.h>
#include <math.h>

// ---------------------------------------------------------------- constants
#define NPTS   32400            // H*W = 135*240
#define CDIM   512              // feature dim
#define CH4    64               // 16B chunks per bf16 feature row (512*2/16)
#define NCLS   19
#define KNN    9
#define FULLN  (1080*1920)

#define BM     128              // rows per workgroup (LDS-resident A)
#define BN     64               // column-chunk width streamed per iteration
#define NJCH   ((NPTS + BN - 1) / BN)   // 507
#define NBLK   ((NPTS + BM - 1) / BM)   // 254
#define BIGF   3.0e38f
#define FMAXV  3.4028235e38f

typedef __attribute__((ext_vector_type(16))) __bf16 bf16x16;
typedef __attribute__((ext_vector_type(8)))  float  f32x8;

union ABFrag { bf16x16 v; uint4 q[2]; };

// ------------------------------------------------------------ prep kernels
// Transpose (C,N) fp32 -> row-major (N,C) bf16 (RNE) + squared norms.
__global__ void prep_convert(const float* __restrict__ feat,
                             unsigned short* __restrict__ Fbf,
                             float* __restrict__ sq)
{
    int n = blockIdx.x * blockDim.x + threadIdx.x;
    if (n >= NPTS) return;
    float s = 0.f;
    #pragma unroll 8
    for (int c = 0; c < CDIM; ++c) {
        float v = feat[(size_t)c * NPTS + n];        // coalesced across n
        s += v * v;
        unsigned int u = __float_as_uint(v);
        unsigned int r = u + 0x7fffu + ((u >> 16) & 1u);   // round-nearest-even
        Fbf[(size_t)n * CDIM + c] = (unsigned short)(r >> 16);
    }
    sq[n] = s;
}

// argmax over class dim (softmax is argmax-invariant); first-max wins like jnp.argmax
__global__ void prep_labels(const float* __restrict__ o, int* __restrict__ labels)
{
    int n = blockIdx.x * blockDim.x + threadIdx.x;
    if (n >= NPTS) return;
    float best = o[n];
    int bi = 0;
    #pragma unroll
    for (int c = 1; c < NCLS; ++c) {
        float v = o[(size_t)c * NPTS + n];
        if (v > best) { best = v; bi = c; }
    }
    labels[n] = bi;
}

__global__ void zero_out(float* __restrict__ out)
{
    int i = blockIdx.x * blockDim.x + threadIdx.x;
    if (i < FULLN) out[i] = 0.f;
}

// -------------------------------------------------------------- top-9 list
// Unsorted 9-entry list + running max; on insert, replace first slot == wmax.
__device__ __forceinline__ void topk_insert(float x, int j,
                                            float (&tv)[KNN], int (&tj)[KNN],
                                            float &wmax)
{
    bool done = false;
    #pragma unroll
    for (int i = 0; i < KNN; ++i) {
        bool m = (!done) && (tv[i] == wmax);
        tv[i] = m ? x : tv[i];
        tj[i] = m ? j : tj[i];
        done = done || m;
    }
    float w = tv[0];
    #pragma unroll
    for (int i = 1; i < KNN; ++i) w = fmaxf(w, tv[i]);
    wmax = w;
}

// ------------------------------------------------------------- main kernel
// One workgroup owns BM=128 rows (A resident in LDS). Sweep all columns in
// BN=64 chunks with register-prefetch pipelining. 8 waves each compute a
// 32x32 output tile (2 row-strips x 2 col-tiles). Fully unrolled k-loop:
// every ds_load_b128 is base + immediate off two LAUNDERED base registers,
// so the loop body is pure ds_load / s_wait_dscnt / v_wmma (no VALU, no
// WMMA->VALU hazard NOPs).
__global__ __launch_bounds__(256, 1)
void knn_entropy_kernel(const uint4* __restrict__ F4,     // [NPTS][CH4] bf16 rows
                        const float* __restrict__ sq,
                        const int*   __restrict__ labels,
                        float* __restrict__ out)
{
    __shared__ uint4 ldsA[BM * CH4];                 // 128 KB : A rows, bf16 row-major
    __shared__ uint4 ldsB[BN * CH4];                 //  64 KB : B cols, bf16 row-major
    __shared__ __align__(16) float ldsSq[BN];        //  sq for current chunk
    __shared__ __align__(16) float ldsD[8 * 256];    //   8 KB : per-wave 16x16 D tile

    const int tid  = threadIdx.x;
    const int wave = tid >> 5;
    const int lane = tid & 31;
    const int l16  = lane & 15;
    const int half = lane >> 4;
    const int rowGroup = wave >> 1;                  // 0..3 -> 32-row group
    const int colGroup = wave & 1;                   // 0..1 -> 32-col group
    const int rowBase  = blockIdx.x * BM;

    // ---- stage A block once (clamped at the tail)
    #pragma unroll
    for (int i = 0; i < (BM * CH4) / 256; ++i) {     // 32 iterations
        int cid = tid + i * 256;
        int r = cid >> 6, o = cid & 63;
        int src = rowBase + r; if (src >= NPTS) src = NPTS - 1;
        ldsA[cid] = F4[src * CH4 + o];
    }

    // per-lane top-9 lists: one per row-strip (s=0,1); this lane covers
    // row = rowGroup*32 + s*16 + l16, columns == colGroup/half comb.
    float tv[2][KNN]; int tj[2][KNN]; float wmax[2] = { FMAXV, FMAXV };
    #pragma unroll
    for (int s = 0; s < 2; ++s)
        #pragma unroll
        for (int i = 0; i < KNN; ++i) { tv[s][i] = FMAXV; tj[s][i] = 0; }

    // ---- laundered per-lane LDS byte offsets: opaque to CSE/addressing
    //      folding, so each becomes ONE loop-invariant base register and all
    //      inner-loop DS accesses use immediate offsets (<= 17376).
    unsigned aOff = (unsigned)(((rowGroup * 32 + l16) * CH4 + half) * 16);
    unsigned bOff = (unsigned)(((colGroup * 32 + l16) * CH4 + half) * 16);
    asm volatile("" : "+v"(aOff), "+v"(bOff));
    const char* const aP = (const char*)ldsA + aOff;
    const char* const bP = (const char*)ldsB + bOff;
    float* const myD = ldsD + wave * 256;

    // ---- register prefetch of B chunk + sq (hides L2 latency behind compute)
    uint4 pb[16];
    float psq = BIGF;
    auto prefetch = [&](int ch) {
        #pragma unroll
        for (int i = 0; i < 16; ++i) {
            int cid = tid + i * 256;
            int c = cid >> 6, o = cid & 63;
            int j = ch * BN + c;
            int src = (j < NPTS) ? j : (NPTS - 1);
            pb[i] = F4[src * CH4 + o];
        }
        if (tid < BN) {
            int j = ch * BN + tid;
            psq = (j < NPTS) ? sq[j] : BIGF;         // pad scores never win
        }
    };
    prefetch(0);

    for (int ch = 0; ch < NJCH; ++ch) {
        __syncthreads();                             // prior chunk fully consumed
        // ---- commit prefetched B chunk to LDS
        #pragma unroll
        for (int i = 0; i < 16; ++i)
            ldsB[tid + i * 256] = pb[i];
        if (tid < BN) ldsSq[tid] = psq;
        if (ch + 1 < NJCH) prefetch(ch + 1);         // issue next chunk's loads
        __syncthreads();

        // ---- k-loop: dot products via bf16 WMMA, f32 accumulate
        f32x8 acc00 = (f32x8){0,0,0,0,0,0,0,0};
        f32x8 acc01 = (f32x8){0,0,0,0,0,0,0,0};
        f32x8 acc10 = (f32x8){0,0,0,0,0,0,0,0};
        f32x8 acc11 = (f32x8){0,0,0,0,0,0,0,0};

        #pragma unroll
        for (int ks = 0; ks < CDIM / 32; ++ks) {
            ABFrag a0, a1, b0, b1;
            a0.q[0] = *(const uint4*)(aP + ks * 64);
            a0.q[1] = *(const uint4*)(aP + ks * 64 + 32);
            a1.q[0] = *(const uint4*)(aP + 16384 + ks * 64);
            a1.q[1] = *(const uint4*)(aP + 16384 + ks * 64 + 32);
            b0.q[0] = *(const uint4*)(bP + ks * 64);
            b0.q[1] = *(const uint4*)(bP + ks * 64 + 32);
            b1.q[0] = *(const uint4*)(bP + 16384 + ks * 64);
            b1.q[1] = *(const uint4*)(bP + 16384 + ks * 64 + 32);
            acc00 = __builtin_amdgcn_wmma_f32_16x16x32_bf16(
                false, a0.v, false, b0.v, (short)0, acc00, false, false);
            acc10 = __builtin_amdgcn_wmma_f32_16x16x32_bf16(
                false, a1.v, false, b0.v, (short)0, acc10, false, false);
            acc01 = __builtin_amdgcn_wmma_f32_16x16x32_bf16(
                false, a0.v, false, b1.v, (short)0, acc01, false, false);
            acc11 = __builtin_amdgcn_wmma_f32_16x16x32_bf16(
                false, a1.v, false, b1.v, (short)0, acc11, false, false);
        }

        // ---- selection: bounce each 16x16 tile through LDS, scan row-major
        f32x8 accs[2][2] = { { acc00, acc01 }, { acc10, acc11 } };
        #pragma unroll
        for (int s = 0; s < 2; ++s) {
            #pragma unroll
            for (int t = 0; t < 2; ++t) {
                #pragma unroll
                for (int v = 0; v < 8; ++v)          // C/D layout: m = v + 8*half
                    myD[(v + 8 * half) * 16 + l16] = accs[s][t][v];
                asm volatile("s_wait_dscnt 0x0" ::: "memory");  // same-wave LDS RAW

                const float4* rp = (const float4*)(myD + l16 * 16 + half * 8);
                float4 d0 = rp[0], d1 = rp[1];
                const float4* qp =
                    (const float4*)(ldsSq + colGroup * 32 + t * 16 + half * 8);
                float4 q0 = qp[0], q1 = qp[1];
                int jb = ch * BN + colGroup * 32 + t * 16 + half * 8;

                float xv[8] = { q0.x - 2.f*d0.x, q0.y - 2.f*d0.y,
                                q0.z - 2.f*d0.z, q0.w - 2.f*d0.w,
                                q1.x - 2.f*d1.x, q1.y - 2.f*d1.y,
                                q1.z - 2.f*d1.z, q1.w - 2.f*d1.w };
                #pragma unroll
                for (int u = 0; u < 8; ++u) {
                    if (xv[u] < wmax[s])
                        topk_insert(xv[u], jb + u, tv[s], tj[s], wmax[s]);
                }
            }
        }
    }

    // ---- cross-wave merge: each row's 4 partial lists (colGroup x half)
    __syncthreads();                                  // ldsB free for reuse
    float* mrg = (float*)ldsB;                        // 128 rows * 4 parts * 18 = 36 KB
    #pragma unroll
    for (int s = 0; s < 2; ++s) {
        int rowLocal = rowGroup * 32 + s * 16 + l16;
        int part = colGroup * 2 + half;
        float* p = mrg + (rowLocal * 4 + part) * 18;
        #pragma unroll
        for (int i = 0; i < KNN; ++i) {
            p[i]     = tv[s][i];
            p[9 + i] = __int_as_float(tj[s][i]);
        }
    }
    __syncthreads();

    if (tid < BM) {
        float fv[KNN]; int fj[KNN]; float fw = FMAXV;
        #pragma unroll
        for (int i = 0; i < KNN; ++i) { fv[i] = FMAXV; fj[i] = 0; }
        for (int part = 0; part < 4; ++part) {
            const float* p = mrg + (tid * 4 + part) * 18;
            #pragma unroll
            for (int i = 0; i < KNN; ++i) {
                float x = p[i];
                int   j = __float_as_int(p[9 + i]);
                if (x < fw) topk_insert(x, j, fv, fj, fw);
            }
        }
        int lab[KNN];
        #pragma unroll
        for (int i = 0; i < KNN; ++i) {
            int j = fj[i];
            j = (j < 0) ? 0 : (j >= NPTS ? NPTS - 1 : j);
            lab[i] = labels[j];
        }
        float ent = 0.f;
        for (int cls = 0; cls < NCLS; ++cls) {
            int c = 0;
            #pragma unroll
            for (int i = 0; i < KNN; ++i) c += (lab[i] == cls) ? 1 : 0;
            float p = (float)c * (1.0f / (float)KNN);
            ent += -p * __logf(p + 1e-6f);
        }
        ent *= (1.0f / 2.9444389791664403f);          // 1/ln(19)
        int row = rowBase + tid;
        if (row < NPTS) out[row] = ent;
    }
}

// ----------------------------------------------------------------- launch
extern "C" void kernel_launch(void* const* d_in, const int* in_sizes, int n_in,
                              void* d_out, int out_size, void* d_ws, size_t ws_size,
                              hipStream_t stream)
{
    const float* feat = (const float*)d_in[0];   // (1, 512, 135, 240) fp32
    const float* outp = (const float*)d_in[1];   // (1, 19, 135, 240)  fp32
    float* out = (float*)d_out;                  // 1080*1920 fp32

    // workspace layout (≈33.4 MB): bf16 features | sq | labels
    char* ws = (char*)d_ws;
    unsigned short* Fbf = (unsigned short*)ws;                           // N*C*2
    float* sq  = (float*)(ws + (size_t)NPTS * CDIM * 2);                 // N*4
    int*   lab = (int*)  (ws + (size_t)NPTS * CDIM * 2 + (size_t)NPTS * 4);

    prep_convert<<<(NPTS + 255) / 256, 256, 0, stream>>>(feat, Fbf, sq);
    prep_labels <<<(NPTS + 255) / 256, 256, 0, stream>>>(outp, lab);
    zero_out    <<<(FULLN + 255) / 256, 256, 0, stream>>>(out);
    knn_entropy_kernel<<<NBLK, 256, 0, stream>>>((const uint4*)Fbf, sq, lab, out);
}
